// MultiHeadLatentAttention_10282151707424
// MI455X (gfx1250) — matrix operand order repaired and checked by reading, not compiled, not run on hardware
//
#include <hip/hip_runtime.h>

#define DEV_INLINE __device__ __forceinline__

typedef __attribute__((ext_vector_type(16))) __bf16 v16bf;
typedef __attribute__((ext_vector_type(8)))  float  v8f;

union Frag { v16bf v; uint4 q[2]; unsigned short u[16]; };
union AccF { v8f v; float f[8]; };
union V128 { uint4 q[4]; unsigned short u[32]; };

DEV_INLINE unsigned short f2bf(float f) {
    unsigned int u = __builtin_bit_cast(unsigned int, f);
    u += 0x7fffu + ((u >> 16) & 1u);          // round-to-nearest-even
    return (unsigned short)(u >> 16);
}
DEV_INLINE float bf2f(unsigned short h) {
    unsigned int u = ((unsigned int)h) << 16;
    return __builtin_bit_cast(float, u);
}
DEV_INLINE v8f wmma_bf16(const Frag& a, const Frag& b, v8f c) {
    return __builtin_amdgcn_wmma_f32_16x16x32_bf16(false, a.v, false, b.v,
                                                   (short)0, c, false, false);
}

// ---------------------------------------------------------------- conversions
__global__ __launch_bounds__(256)
void k_f32_to_bf16(const float* __restrict__ s, unsigned short* __restrict__ d, long n) {
    long i = (long)blockIdx.x * blockDim.x + threadIdx.x;
    long st = (long)gridDim.x * blockDim.x;
    for (; i < n; i += st) d[i] = f2bf(s[i]);
}

// weight convert + transpose + K-pad: src f32 [K x N] -> dst bf16 [N x Kpad]
__global__ __launch_bounds__(256)
void k_convert_w_t(const float* __restrict__ s, unsigned short* __restrict__ d,
                   int K, int N, int Kpad, long n) {   // n = N*Kpad
    long i = (long)blockIdx.x * blockDim.x + threadIdx.x;
    long st = (long)gridDim.x * blockDim.x;
    for (; i < n; i += st) {
        int col = (int)(i / Kpad);     // output row = original column
        int k   = (int)(i % Kpad);
        d[i] = (k < K) ? f2bf(s[(size_t)k * N + col]) : (unsigned short)0;
    }
}

// ---------------------------------------------------------------- WMMA GEMM
// C[M x N] (f32 or bf16) = A_bf16[M x K, lda] @ Bt_bf16[N x K, ldb (k-stride)]
// grid = (ceil(N/128), M/128), block = 256 (8 waves, 4x2 wave grid),
// wave tile 32x64, K % 32 == 0, lda/ldb % 8 == 0.
__global__ __launch_bounds__(256)
void k_gemm_bf16(const unsigned short* __restrict__ A, int lda,
                 const unsigned short* __restrict__ Bt, int ldb,
                 void* __restrict__ C, int ldc,
                 int N, int K, int store_bf16) {
    __shared__ unsigned short sA[128 * 40];   // [row][k]
    __shared__ unsigned short sB[128 * 40];   // [col][k]

    const int tid  = threadIdx.x;
    const int lane = tid & 31;
    const int wid  = tid >> 5;
    const int hl   = lane >> 4;
    const int l16  = lane & 15;
    const int row0 = blockIdx.y * 128;
    const int col0 = blockIdx.x * 128;
    const int wrow = (wid >> 1) * 32;
    const int wcol = (wid & 1) * 64;

    AccF acc[2][4];
#pragma unroll
    for (int i = 0; i < 2; ++i)
#pragma unroll
        for (int j = 0; j < 4; ++j)
#pragma unroll
            for (int r = 0; r < 8; ++r) acc[i][j].f[r] = 0.f;

    for (int k0 = 0; k0 < K; k0 += 32) {
        {   // A: 128 rows x 32 halves; thread t -> row t>>1, 16-half segment
            int r = tid >> 1, s = (tid & 1) * 16;
            const uint4* src = (const uint4*)(A + (size_t)(row0 + r) * lda + k0 + s);
            uint4 x0 = src[0], x1 = src[1];
            uint4* dst = (uint4*)(sA + r * 40 + s);
            dst[0] = x0; dst[1] = x1;
            if (k0 + 32 < K)
                __builtin_prefetch(A + (size_t)(row0 + r) * lda + k0 + 32 + s, 0, 1);
        }
        {   // Bt: 128 cols x 32 halves; identical pattern, bounds-checked rows
            int c = tid >> 1, s = (tid & 1) * 16;
            uint4 x0 = {0, 0, 0, 0}, x1 = {0, 0, 0, 0};
            if (col0 + c < N) {
                const uint4* src = (const uint4*)(Bt + (size_t)(col0 + c) * ldb + k0 + s);
                x0 = src[0]; x1 = src[1];
                if (k0 + 32 < K)
                    __builtin_prefetch(Bt + (size_t)(col0 + c) * ldb + k0 + 32 + s, 0, 1);
            }
            uint4* dst = (uint4*)(sB + c * 40 + s);
            dst[0] = x0; dst[1] = x1;
        }
        __syncthreads();

        Frag a[2], b[4];
#pragma unroll
        for (int mr = 0; mr < 2; ++mr) {
            const unsigned short* p = sA + (wrow + mr * 16 + l16) * 40;
            a[mr].q[0] = *(const uint4*)(p + hl * 8);
            a[mr].q[1] = *(const uint4*)(p + 16 + hl * 8);
        }
#pragma unroll
        for (int nc = 0; nc < 4; ++nc) {
            const unsigned short* p = sB + (wcol + nc * 16 + l16) * 40;
            b[nc].q[0] = *(const uint4*)(p + hl * 8);
            b[nc].q[1] = *(const uint4*)(p + 16 + hl * 8);
        }
#pragma unroll
        for (int mr = 0; mr < 2; ++mr)
#pragma unroll
            for (int nc = 0; nc < 4; ++nc)
                acc[mr][nc].v = wmma_bf16(a[mr], b[nc], acc[mr][nc].v);
        __syncthreads();
    }

#pragma unroll
    for (int mr = 0; mr < 2; ++mr)
#pragma unroll
        for (int nc = 0; nc < 4; ++nc)
#pragma unroll
            for (int r = 0; r < 8; ++r) {
                int grow = row0 + wrow + mr * 16 + r + 8 * hl;
                int gcol = col0 + wcol + nc * 16 + l16;
                if (gcol < N) {
                    float v = acc[mr][nc].f[r];
                    if (store_bf16)
                        ((unsigned short*)C)[(size_t)grow * ldc + gcol] = f2bf(v);
                    else
                        ((float*)C)[(size_t)grow * ldc + gcol] = v;
                }
            }
}

// ---------------------------------------------------------------- RMSNorm
__global__ __launch_bounds__(256)
void k_rmsnorm_bf16(const float* __restrict__ x, int ldx, int cols,
                    const float* __restrict__ g,
                    unsigned short* __restrict__ y, int ldy, int ycols) {
    __shared__ float red[256];
    const int row = blockIdx.x;
    const float* xr = x + (size_t)row * ldx;
    float ss = 0.f;
    for (int c = threadIdx.x; c < cols; c += 256) { float v = xr[c]; ss += v * v; }
    red[threadIdx.x] = ss;
    __syncthreads();
    for (int s = 128; s > 0; s >>= 1) {
        if (threadIdx.x < s) red[threadIdx.x] += red[threadIdx.x + s];
        __syncthreads();
    }
    float inv = rsqrtf(red[0] / (float)cols + 1e-5f);
    for (int c = threadIdx.x; c < ycols; c += 256) {
        float v = (c < cols) ? xr[c] * inv * g[c] : 0.f;
        y[(size_t)row * ldy + c] = f2bf(v);
    }
}

// ---------------------------------------------------------------- RoPE (Q)
__global__ __launch_bounds__(256)
void k_rope_q(const unsigned short* __restrict__ Qb, unsigned short* __restrict__ Qh, int S) {
    const int row = blockIdx.x;
    const int pos = row % S;
    const unsigned short* src = Qb + (size_t)row * 2048;
    unsigned short* dst = Qh + (size_t)row * 2048;
    for (int pid = threadIdx.x; pid < 1024; pid += blockDim.x) {
        int h = pid >> 6, j = pid & 63;
        if (j < 32) {                       // nope half: passthrough
            int c = h * 128 + 2 * j;
            dst[c] = src[c]; dst[c + 1] = src[c + 1];
        } else {                            // rope half: interleaved rotation
            int jj = j - 32;
            int c = h * 128 + 64 + 2 * jj;
            float ang = (float)pos * __powf(10000.0f, -(float)jj * (1.0f / 64.0f));
            float cs = __cosf(ang), sn = __sinf(ang);
            float x0 = bf2f(src[c]), x1 = bf2f(src[c + 1]);
            dst[c]     = f2bf(x0 * cs - x1 * sn);
            dst[c + 1] = f2bf(x1 * cs + x0 * sn);
        }
    }
}

// ------------------------------------------------------- RoPE (K, broadcast)
__global__ __launch_bounds__(256)
void k_rope_k(const float* __restrict__ ckv, int ldc, int kvp,
              unsigned short* __restrict__ Kh, int S) {
    const int row = blockIdx.x;
    const int pos = row % S;
    const float* src = ckv + (size_t)row * ldc + kvp;
    unsigned short* dst = Kh + (size_t)row * 2048;
    for (int t = threadIdx.x; t < 512; t += blockDim.x) {
        int h = t >> 5, jj = t & 31;
        float ang = (float)pos * __powf(10000.0f, -(float)jj * (1.0f / 64.0f));
        float cs = __cosf(ang), sn = __sinf(ang);
        float x0 = src[2 * jj], x1 = src[2 * jj + 1];
        int c = h * 128 + 64 + 2 * jj;
        dst[c]     = f2bf(x0 * cs - x1 * sn);
        dst[c + 1] = f2bf(x1 * cs + x0 * sn);
    }
}

// -------------------------------------------------- split KV -> K_nope / V
__global__ __launch_bounds__(256)
void k_scatter_kv(const unsigned short* __restrict__ KVb,
                  unsigned short* __restrict__ Kh,
                  unsigned short* __restrict__ Vh) {
    const size_t row = blockIdx.x;
    const unsigned short* src = KVb + row * 3072;
    unsigned short* kd = Kh + row * 2048;
    unsigned short* vd = Vh + row * 2048;
    for (int c = threadIdx.x; c < 3072; c += blockDim.x) {
        int h = c / 192, d = c % 192;
        unsigned short v = src[c];
        if (d < 64) kd[h * 128 + d] = v;
        else        vd[h * 128 + (d - 64)] = v;
    }
}

// -------------------------------------------------- flash attention (causal)
// grid = (S/64, H, B), block = 128 (4 waves x 16 query rows). dh = 128.
__global__ __launch_bounds__(128)
void k_mla_attention(const unsigned short* __restrict__ Qh,
                     const unsigned short* __restrict__ Kh,
                     const unsigned short* __restrict__ Vh,
                     unsigned short* __restrict__ O,
                     int S, int H) {
    __shared__ unsigned short sQ[64 * 136];   // [qrow][d]
    __shared__ unsigned short sK[32 * 136];   // [key][d]
    __shared__ unsigned short sV[128 * 40];   // transposed: [d][key]
    __shared__ unsigned short sP[4 * 16 * 40];// per-wave P [row][key]

    const int tid  = threadIdx.x;
    const int lane = tid & 31;
    const int w    = tid >> 5;
    const int hl   = lane >> 4;
    const int l16  = lane & 15;

    const int qt = blockIdx.x;
    const int h  = blockIdx.y;
    const int b  = blockIdx.z;
    const int q0 = qt * 64;
    const size_t rowbase = (size_t)b * S;
    const int colbase = h * 128;
    const int ld = H * 128;    // 2048

    {   // load Q tile 64x128
        int r  = tid >> 1;
        int s0 = (tid & 1) * 64;
        const uint4* src = (const uint4*)(Qh + (rowbase + q0 + r) * ld + colbase + s0);
        uint4* dst = (uint4*)(sQ + r * 136 + s0);
#pragma unroll
        for (int i = 0; i < 8; ++i) dst[i] = src[i];
    }

    AccF o[8];
#pragma unroll
    for (int dc = 0; dc < 8; ++dc)
#pragma unroll
        for (int r = 0; r < 8; ++r) o[dc].f[r] = 0.f;
    float mrow[8], lrow[8];
#pragma unroll
    for (int r = 0; r < 8; ++r) { mrow[r] = -1e30f; lrow[r] = 0.f; }

    const float scale = 0.08838834764831845f;  // 1/sqrt(128)
    const int kend = q0 + 63;

    for (int k0 = 0; k0 <= kend; k0 += 32) {
        __syncthreads();
        {   // load K chunk (row-major) and V chunk (transposed)
            int kr  = tid >> 2;
            int seg = (tid & 3) * 32;
            const uint4* srcK = (const uint4*)(Kh + (rowbase + k0 + kr) * ld + colbase + seg);
            uint4* dstK = (uint4*)(sK + kr * 136 + seg);
#pragma unroll
            for (int i = 0; i < 4; ++i) dstK[i] = srcK[i];
            V128 vv;
            const uint4* srcV = (const uint4*)(Vh + (rowbase + k0 + kr) * ld + colbase + seg);
#pragma unroll
            for (int i = 0; i < 4; ++i) vv.q[i] = srcV[i];
#pragma unroll
            for (int i = 0; i < 32; ++i) sV[(seg + i) * 40 + kr] = vv.u[i];
        }
        __syncthreads();

        // S = Q @ K^T  (two 16x16 key tiles, K-dim looped 4 x 32)
        AccF s0, s1;
#pragma unroll
        for (int r = 0; r < 8; ++r) { s0.f[r] = 0.f; s1.f[r] = 0.f; }
#pragma unroll
        for (int dc = 0; dc < 4; ++dc) {
            Frag a, b0, b1;
            const unsigned short* pa = sQ + (w * 16 + l16) * 136 + dc * 32;
            a.q[0] = *(const uint4*)(pa + hl * 8);
            a.q[1] = *(const uint4*)(pa + 16 + hl * 8);
            const unsigned short* pb0 = sK + l16 * 136 + dc * 32;
            b0.q[0] = *(const uint4*)(pb0 + hl * 8);
            b0.q[1] = *(const uint4*)(pb0 + 16 + hl * 8);
            const unsigned short* pb1 = sK + (16 + l16) * 136 + dc * 32;
            b1.q[0] = *(const uint4*)(pb1 + hl * 8);
            b1.q[1] = *(const uint4*)(pb1 + 16 + hl * 8);
            s0.v = wmma_bf16(a, b0, s0.v);
            s1.v = wmma_bf16(a, b1, s1.v);
        }

        // online softmax (rows live in D-layout: row = r + 8*hl, col = l16)
#pragma unroll
        for (int r = 0; r < 8; ++r) {
            int qg = q0 + w * 16 + r + 8 * hl;
            float v0 = s0.f[r] * scale; if (k0 + l16 > qg)      v0 = -1e30f;
            float v1 = s1.f[r] * scale; if (k0 + 16 + l16 > qg) v1 = -1e30f;
            float mx = fmaxf(v0, v1);
#pragma unroll
            for (int m = 1; m < 16; m <<= 1) mx = fmaxf(mx, __shfl_xor(mx, m, 32));
            float mnew  = fmaxf(mrow[r], mx);
            float alpha = __expf(mrow[r] - mnew);
            float p0 = __expf(v0 - mnew), p1 = __expf(v1 - mnew);
            float ps = p0 + p1;
#pragma unroll
            for (int m = 1; m < 16; m <<= 1) ps += __shfl_xor(ps, m, 32);
            lrow[r] = lrow[r] * alpha + ps;
            mrow[r] = mnew;
#pragma unroll
            for (int dc = 0; dc < 8; ++dc) o[dc].f[r] *= alpha;
            sP[(w * 16 + r + 8 * hl) * 40 + l16]      = f2bf(p0);
            sP[(w * 16 + r + 8 * hl) * 40 + 16 + l16] = f2bf(p1);
        }
        __syncthreads();

        // O += P @ V  (8 d-chunks of 16)
#pragma unroll
        for (int dc = 0; dc < 8; ++dc) {
            Frag a, bb;
            const unsigned short* pa = sP + (w * 16 + l16) * 40;
            a.q[0] = *(const uint4*)(pa + hl * 8);
            a.q[1] = *(const uint4*)(pa + 16 + hl * 8);
            const unsigned short* pv = sV + (dc * 16 + l16) * 40;
            bb.q[0] = *(const uint4*)(pv + hl * 8);
            bb.q[1] = *(const uint4*)(pv + 16 + hl * 8);
            o[dc].v = wmma_bf16(a, bb, o[dc].v);
        }
    }

#pragma unroll
    for (int dc = 0; dc < 8; ++dc)
#pragma unroll
        for (int r = 0; r < 8; ++r) {
            int qg = q0 + w * 16 + r + 8 * hl;
            float v = o[dc].f[r] / lrow[r];
            O[(rowbase + qg) * ld + colbase + dc * 16 + l16] = f2bf(v);
        }
}

// ================================================================ launcher
extern "C" void kernel_launch(void* const* d_in, const int* in_sizes, int n_in,
                              void* d_out, int out_size, void* d_ws, size_t ws_size,
                              hipStream_t stream) {
    const float* x    = (const float*)d_in[0];
    const float* Wqd  = (const float*)d_in[1];
    const float* Wqu  = (const float*)d_in[2];
    const float* Wkvd = (const float*)d_in[3];
    const float* Wkvu = (const float*)d_in[4];
    const float* Wo   = (const float*)d_in[5];
    const float* gq   = (const float*)d_in[6];
    const float* gkv  = (const float*)d_in[7];

    const int B = 2, S = 2048, D = 2048, H = 16;
    const int R = B * S;                    // 4096 token rows
    const int QP = 1024;                    // q_proj
    const int KVP = 1365, ROPE = 64;        // kv_proj, rope
    const int KVC = KVP + ROPE;             // 1429
    const int KVPAD = 1376;                 // kv_proj padded to mult-of-32
    const int KVUP = 3072;                  // D + H*nope

    char* ws = (char*)d_ws;
    size_t off = 0;
    auto carve = [&](size_t bytes) -> void* {
        void* p = ws + off;
        off = (off + bytes + 255) & ~(size_t)255;
        return p;
    };
    unsigned short* Xb    = (unsigned short*)carve((size_t)R * D * 2);
    // transposed weights: [N][Kpad], k contiguous
    unsigned short* Wqd_t = (unsigned short*)carve((size_t)QP * D * 2);      // [1024][2048]
    unsigned short* Wqu_t = (unsigned short*)carve((size_t)D * QP * 2);      // [2048][1024]
    unsigned short* Wkd_t = (unsigned short*)carve((size_t)KVC * D * 2);     // [1429][2048]
    unsigned short* Wku_t = (unsigned short*)carve((size_t)KVUP * KVPAD * 2);// [3072][1376]
    unsigned short* Wo_t  = (unsigned short*)carve((size_t)D * D * 2);       // [2048][2048]
    float*          cqraw = (float*)carve((size_t)R * QP * 4);
    unsigned short* cq_b  = (unsigned short*)carve((size_t)R * QP * 2);
    unsigned short* Qb    = (unsigned short*)carve((size_t)R * D * 2);
    unsigned short* KVl_b = (unsigned short*)carve((size_t)R * KVPAD * 2);
    unsigned short* KVb   = (unsigned short*)carve((size_t)R * KVUP * 2);
    unsigned short* Qh    = (unsigned short*)carve((size_t)R * D * 2);
    unsigned short* Kh    = (unsigned short*)carve((size_t)R * D * 2);
    unsigned short* Vh    = (unsigned short*)carve((size_t)R * D * 2);
    unsigned short* Attn  = (unsigned short*)carve((size_t)R * D * 2);

    float* out = (float*)d_out;
    float* ckv = out + (size_t)R * D;       // compressed_kv output region

    // --- bf16 conversions (weights transposed + K-padded)
    k_f32_to_bf16<<<2048, 256, 0, stream>>>(x, Xb, (long)R * D);
    k_convert_w_t<<<1024, 256, 0, stream>>>(Wqd,  Wqd_t, D,   QP,   D,     (long)QP * D);
    k_convert_w_t<<<1024, 256, 0, stream>>>(Wqu,  Wqu_t, QP,  D,    QP,    (long)D * QP);
    k_convert_w_t<<<1024, 256, 0, stream>>>(Wkvd, Wkd_t, D,   KVC,  D,     (long)KVC * D);
    k_convert_w_t<<<1024, 256, 0, stream>>>(Wkvu, Wku_t, KVP, KVUP, KVPAD, (long)KVUP * KVPAD);
    k_convert_w_t<<<1024, 256, 0, stream>>>(Wo,   Wo_t,  D,   D,    D,     (long)D * D);

    // --- cq_raw = x @ Wq_down (f32 out for RMSNorm precision)
    k_gemm_bf16<<<dim3(QP / 128, R / 128), 256, 0, stream>>>(
        Xb, D, Wqd_t, D, cqraw, QP, QP, D, 0);
    // --- compressed_kv = x @ Wkv_down -> straight into d_out region 2
    k_gemm_bf16<<<dim3((KVC + 127) / 128, R / 128), 256, 0, stream>>>(
        Xb, D, Wkd_t, D, ckv, KVC, KVC, D, 0);

    // --- cq = rmsnorm(cq_raw, gq) -> bf16
    k_rmsnorm_bf16<<<R, 256, 0, stream>>>(cqraw, QP, QP, gq, cq_b, QP, QP);
    // --- Q = cq @ Wq_up (bf16 out)
    k_gemm_bf16<<<dim3(D / 128, R / 128), 256, 0, stream>>>(
        cq_b, QP, Wqu_t, QP, Qb, D, D, QP, 1);
    // --- RoPE on Q heads -> Qh
    k_rope_q<<<R, 256, 0, stream>>>(Qb, Qh, S);

    // --- KV_lora = rmsnorm(ckv[:, :kvp], gkv) -> bf16 (K padded with zeros)
    k_rmsnorm_bf16<<<R, 256, 0, stream>>>(ckv, KVC, KVP, gkv, KVl_b, KVPAD, KVPAD);
    // --- K_rope = rope(ckv[:, kvp:]) broadcast across heads -> Kh
    k_rope_k<<<R, 256, 0, stream>>>(ckv, KVC, KVP, Kh, S);
    // --- KV = KV_lora @ Wkv_up (bf16 out)
    k_gemm_bf16<<<dim3(KVUP / 128, R / 128), 256, 0, stream>>>(
        KVl_b, KVPAD, Wku_t, KVPAD, KVb, KVUP, KVUP, KVPAD, 1);
    // --- split KV -> K_nope (Kh) and V (Vh)
    k_scatter_kv<<<R, 256, 0, stream>>>(KVb, Kh, Vh);

    // --- causal flash attention per (b, h, 64-row query tile)
    k_mla_attention<<<dim3(S / 64, H, B), 128, 0, stream>>>(Qh, Kh, Vh, Attn, S, H);

    // --- out = attn @ Wo (f32 into d_out region 1)
    k_gemm_bf16<<<dim3(D / 128, R / 128), 256, 0, stream>>>(
        Attn, D, Wo_t, D, out, D, D, D, 0);
}
